// KOBE_77206332113784
// MI455X (gfx1250) — compile-verified, matrix operand order checked by reading, not run on hardware
//
#include <hip/hip_runtime.h>
#include <hip/hip_bf16.h>

typedef __attribute__((ext_vector_type(16))) _Float16 v16h;
typedef __attribute__((ext_vector_type(8)))  float    v8f;

#define NBITS 32
#define TPAD  36   // padded LDS row stride (floats): 36*m mod 64 distinct for m=0..15,
                   // and 36*4 bytes is 16B-aligned -> conflict-free float4 gathers

// Build symmetric pair matrix M (f16, 32x32, entries k/2) and linear vector L (f32, 32)
__global__ __launch_bounds__(1024) void KOBE_build_tables(
    const float* __restrict__ kern, const int* __restrict__ pidx,
    _Float16* __restrict__ M, float* __restrict__ L, int num_terms) {
  int t = threadIdx.x;
  if (t < NBITS * NBITS) M[t] = (_Float16)0.0f;
  if (t < NBITS)         L[t] = 0.0f;
  __syncthreads();
  for (int u = t; u < num_terms; u += blockDim.x) {
    int i = pidx[2 * u + 0];
    int j = pidx[2 * u + 1];
    float k = kern[u];
    if (j >= NBITS) {                       // singleton term (i, pad)
      L[i] = k;
    } else {                                // pair term i<j
      _Float16 h = (_Float16)(k * 0.5f);
      M[i * NBITS + j] = h;
      M[j * NBITS + i] = h;
    }
  }
}

// out[b] = sum_i S[b,i] * ((S·M)[b,i] + L[i]),  S in {±1}
// Per wave tile (16 batch rows): A = 16x32 f16 spins,
// D0 = A·M[:,0:16] + L[0:16], D1 = A·M[:,16:32] + L[16:32]  (2x v_wmma_f32_16x16x32_f16),
// then a per-wave LDS transpose + in-register row-dot.
__global__ __launch_bounds__(256) void KOBE_parity_qform(
    const int* __restrict__ bits, const _Float16* __restrict__ Mw,
    const float* __restrict__ Lw, float* __restrict__ out,
    int batch, int numTiles) {
  __shared__ __align__(16) float sT[8][16 * TPAD];   // 18 KB: per-wave T' tiles

  const int lane = threadIdx.x & 31;
  const int wave = threadIdx.x >> 5;
  const int m    = lane & 15;   // A row / B,C,D column
  const int half = lane >> 4;   // K-half (A,B) / M-half (C,D)

  // ---- loop-invariant operands ----
  // B from symmetric M: B[k][n] = M[k][n] = M[n][k] -> contiguous row reads
  v16h b0 = *(const v16h*)(Mw + m * NBITS + half * 16);
  v16h b1 = *(const v16h*)(Mw + (16 + m) * NBITS + half * 16);
  // C: broadcast linear coefficients per output column
  const float l0 = Lw[m];
  const float l1 = Lw[16 + m];
  v8f c0, c1;
#pragma unroll
  for (int r = 0; r < 8; ++r) { c0[r] = l0; c1[r] = l1; }

  float* sTw = &sT[wave][0];
  const int waveId     = blockIdx.x * 8 + wave;
  const int waveStride = gridDim.x * 8;

  for (int t = waveId; t < numTiles; t += waveStride) {
    const int row  = t * 16 + m;
    const int rowc = row < batch ? row : (batch - 1);
    const int* rp  = bits + (size_t)rowc * NBITS + half * 16;
    int4 w0 = *(const int4*)(rp + 0);
    int4 w1 = *(const int4*)(rp + 4);
    int4 w2 = *(const int4*)(rp + 8);
    int4 w3 = *(const int4*)(rp + 12);

    // prefetch next tile's bits (global_prefetch_b8)
    const int tn = t + waveStride;
    if (tn < numTiles)
      __builtin_prefetch(bits + (size_t)(tn * 16 + m) * NBITS + half * 16, 0, 3);

    const int bv[16] = {w0.x, w0.y, w0.z, w0.w, w1.x, w1.y, w1.z, w1.w,
                        w2.x, w2.y, w2.z, w2.w, w3.x, w3.y, w3.z, w3.w};
    // A (contiguous k order, same as B: lanes 0-15 K=0..15, lanes 16-31 K=16..31);
    // keep f32 spins in registers for the final row-dot (same lane needs same elems)
    v16h a;
    float sreg[16];
#pragma unroll
    for (int e = 0; e < 16; ++e) {
      float s = 1.0f - 2.0f * (float)bv[e];
      sreg[e] = s;
      a[e] = (_Float16)s;
    }

    v8f d0 = __builtin_amdgcn_wmma_f32_16x16x32_f16(false, a, false, b0,
                                                    (short)0, c0, false, false);
    v8f d1 = __builtin_amdgcn_wmma_f32_16x16x32_f16(false, a, false, b1,
                                                    (short)0, c1, false, false);

    // D layout: lanes 0-15 => N=lane, M=r; lanes 16-31 => N=lane-16, M=r+8
#pragma unroll
    for (int r = 0; r < 8; ++r) {
      sTw[(half * 8 + r) * TPAD + m]      = d0[r];
      sTw[(half * 8 + r) * TPAD + 16 + m] = d1[r];
    }
    __builtin_amdgcn_wave_barrier();   // per-wave LDS is hardware in-order; just fence scheduling

    // gather own row (conflict-free via TPAD), dot with in-register spins
    const float4* rowT = (const float4*)(sTw + m * TPAD + half * 16);
    float4 t0 = rowT[0], t1 = rowT[1], t2 = rowT[2], t3 = rowT[3];
    float acc = 0.0f;
    acc += t0.x * sreg[0]  + t0.y * sreg[1]  + t0.z * sreg[2]  + t0.w * sreg[3];
    acc += t1.x * sreg[4]  + t1.y * sreg[5]  + t1.z * sreg[6]  + t1.w * sreg[7];
    acc += t2.x * sreg[8]  + t2.y * sreg[9]  + t2.z * sreg[10] + t2.w * sreg[11];
    acc += t3.x * sreg[12] + t3.y * sreg[13] + t3.z * sreg[14] + t3.w * sreg[15];
    acc += __shfl_xor(acc, 16, 32);
    if (half == 0 && row < batch) out[row] = acc;
    __builtin_amdgcn_wave_barrier();   // keep next iter's stores after this iter's loads
  }
}

extern "C" void kernel_launch(void* const* d_in, const int* in_sizes, int n_in,
                              void* d_out, int out_size, void* d_ws, size_t ws_size,
                              hipStream_t stream) {
  const int*   bits = (const int*)d_in[0];     // (BATCH, 32) int32
  const float* kern = (const float*)d_in[1];   // (528,) f32
  const int*   pidx = (const int*)d_in[2];     // (528, 2) int32
  float* out = (float*)d_out;                  // (BATCH,) f32

  const int num_terms = in_sizes[1];
  const int batch     = in_sizes[0] / NBITS;

  _Float16* M = (_Float16*)d_ws;                    // 32x32 f16 = 2 KB
  float*    L = (float*)((char*)d_ws + 2048);       // 32 f32   = 128 B

  KOBE_build_tables<<<1, 1024, 0, stream>>>(kern, pidx, M, L, num_terms);

  const int numTiles = (batch + 15) / 16;           // 16 batch rows per wave-tile
  int blocks = (numTiles + 15) / 16;                // 8 waves/block, ~2 tiles per wave
  if (blocks < 1) blocks = 1;
  KOBE_parity_qform<<<blocks, 256, 0, stream>>>(bits, M, L, out, batch, numTiles);
}
